// CryptoMamba_17171279249467
// MI455X (gfx1250) — matrix operand order repaired and checked by reading, not compiled, not run on hardware
//
#include <hip/hip_runtime.h>
#include <math.h>

// ---------------------------------------------------------------------------
// CryptoMamba fused pipeline for gfx1250 (MI455X).
//   Dense math  : v_wmma_f32_16x16x32_bf16 (bf16 operands, f32 accum)
//   SSM scan    : fp32 VALU, one workgroup per sequence, state in registers
// Workspace layout (bytes), total ~334 MB:
//   hres  f32  [115200 x 128]  @ 0
//   xn    bf16 [115200 x 128]  @ 58982400
//   xz    bf16 [115200 x 512]  @ 88473600   (xc half later overwritten by y)
//   xc    bf16 [115200 x 256]  @ 206438400
//   xdb   f32  [115200 x  40]  @ 265420800
//   xe    bf16 [115200 x 288]  @ 283852800  (x cast to bf16, zero-padded K)
// ---------------------------------------------------------------------------

#define S_SEQ   480
#define L_SEQ   240
#define F_IN    258
#define F_PAD   288              // 258 padded up to 9 k-tiles of 32
#define DMODEL  128
#define DINNER  256
#define DSTATE  16
#define DTRANK  8
#define DCONV   4
#define DXDB    40
#define NLAYERS 4
#define M_ROWS  (S_SEQ * L_SEQ)  // 115200 rows, divisible by 128

typedef __attribute__((ext_vector_type(16))) __bf16 v16bf;
typedef __attribute__((ext_vector_type(8)))  __bf16 v8bf;
typedef __attribute__((ext_vector_type(8)))  float  v8f;

static __device__ __forceinline__ __bf16 f32_to_bf16(float f) {
  unsigned int u = __builtin_bit_cast(unsigned int, f);
  u += 0x7FFFu + ((u >> 16) & 1u);               // round-to-nearest-even
  unsigned short h = (unsigned short)(u >> 16);
  return __builtin_bit_cast(__bf16, h);
}
static __device__ __forceinline__ float bf16_to_f32(__bf16 b) {
  unsigned int u = ((unsigned int)__builtin_bit_cast(unsigned short, b)) << 16;
  return __builtin_bit_cast(float, u);
}
static __device__ __forceinline__ __bf16 bf16_zero() {
  return __builtin_bit_cast(__bf16, (unsigned short)0);
}

// ---------------------------------------------------------------------------
// Cast x (fp32, 258-wide rows) -> bf16 zero-padded to 288-wide rows.
// One block per row; pure bandwidth.
// ---------------------------------------------------------------------------
__global__ void __launch_bounds__(256)
cast_pad_kernel(const float* __restrict__ x, __bf16* __restrict__ xe)
{
  const size_t row = blockIdx.x;
  const float*  src = x  + row * F_IN;
  __bf16*       dst = xe + row * F_PAD;
  for (int col = threadIdx.x; col < F_PAD; col += 256)
    dst[col] = (col < F_IN) ? f32_to_bf16(src[col]) : bf16_zero();
}

// ---------------------------------------------------------------------------
// Tiled WMMA GEMM:  Out[M x N] (+)= A[M x Kpad] * B[KB x N] (+ bias)
//   A is bf16 with ldA a multiple of 32 and zero-padded to Kpad = ktiles*32,
//   so the A fragment is always two unguarded 16-byte loads.
//   block = 256 threads (8 waves). Each wave owns a 16(M) x 64(N) tile
//   -> block tile 128 x 64; grid = (M/128, Ncompute/64).
//   B (fp32 weights) is converted to bf16 into LDS in exact WMMA-fragment
//   order: Bl[kt][nsub][lane][elem], elem i of lane L maps to
//   (k = kt*32 + (L>>4)*8 + (i>>3)*16 + (i&7), n = nsub*16 + (L&15))  [ISA 7.12.2].
//   A fragment per the 16-bit 16x32 layout: lane L row = L&15,
//   k(i) = kt*32 + (L>>4)*8 + (i>>3)*16 + (i&7)  -> two contiguous runs of 8.
// OUT_MODE: 0 = f32 store, 1 = f32 accumulate, 2 = bf16 store
// ---------------------------------------------------------------------------
template<int OUT_MODE, bool HAS_BIAS>
__global__ void __launch_bounds__(256)
gemm_wmma_kernel(const __bf16* __restrict__ Aptr, const float* __restrict__ Bptr,
                 void* Outp, const float* __restrict__ bias,
                 int ktiles, int KB, int ldA, int ldB, int NB,
                 int ldOut, int Nguard)
{
  __shared__ __align__(32) __bf16 Bl[9 * 4 * 32 * 16];   // up to Kpad=288: 36864 B

  const int tid  = threadIdx.x;
  const int wave = tid >> 5;
  const int lane = tid & 31;
  const int colbase = blockIdx.y * 64;

  // Cooperative B fill (fp32 -> bf16) into fragment order; guard real K and N.
  const int totB = ktiles * 2048;
  for (int e = tid; e < totB; e += 256) {
    int i  = e & 15;
    int ln = (e >> 4) & 31;
    int ns = (e >> 9) & 3;
    int kt = e >> 11;
    int k  = kt * 32 + ((ln >> 4) << 3) + ((i >> 3) << 4) + (i & 7);
    int n  = colbase + ns * 16 + (ln & 15);
    __bf16 v = bf16_zero();
    if (k < KB && n < NB) v = f32_to_bf16(Bptr[(size_t)k * ldB + n]);
    Bl[e] = v;
  }
  __syncthreads();

  const int row = blockIdx.x * 128 + wave * 16 + (lane & 15);
  const size_t rowOff = (size_t)row * (size_t)ldA;
  const int h8 = (lane >> 4) << 3;

  v8f acc0 = {}, acc1 = {}, acc2 = {}, acc3 = {};

  for (int kt = 0; kt < ktiles; ++kt) {
    const int k0 = kt * 32 + h8;
    v16bf a;
    ((v8bf*)&a)[0] = *(const v8bf*)(Aptr + rowOff + k0);        // 16B, aligned
    ((v8bf*)&a)[1] = *(const v8bf*)(Aptr + rowOff + k0 + 16);   // 16B, aligned
    const __bf16* bb = &Bl[(kt * 4) * 512 + lane * 16];
    v16bf b0 = *(const v16bf*)(bb + 0 * 512);
    v16bf b1 = *(const v16bf*)(bb + 1 * 512);
    v16bf b2 = *(const v16bf*)(bb + 2 * 512);
    v16bf b3 = *(const v16bf*)(bb + 3 * 512);
    acc0 = __builtin_amdgcn_wmma_f32_16x16x32_bf16(false, a, false, b0, (short)0, acc0, false, false);
    acc1 = __builtin_amdgcn_wmma_f32_16x16x32_bf16(false, a, false, b1, (short)0, acc1, false, false);
    acc2 = __builtin_amdgcn_wmma_f32_16x16x32_bf16(false, a, false, b2, (short)0, acc2, false, false);
    acc3 = __builtin_amdgcn_wmma_f32_16x16x32_bf16(false, a, false, b3, (short)0, acc3, false, false);
  }

  // Epilogue.  C/D layout: VGPR v -> (M = v + 8*(lane>>4), N = lane&15).
  const int rowb = blockIdx.x * 128 + wave * 16 + ((lane >> 4) << 3);
  const int colw = lane & 15;
  #define EPI(JJ, ACC)                                                        \
    { int col = colbase + (JJ) * 16 + colw;                                   \
      if (col < Nguard) {                                                     \
        _Pragma("unroll")                                                     \
        for (int v = 0; v < 8; ++v) {                                         \
          size_t o = (size_t)(rowb + v) * (size_t)ldOut + col;                \
          float val = ACC[v];                                                 \
          if (HAS_BIAS) val += bias[col];                                     \
          if (OUT_MODE == 0)      ((float*)Outp)[o]  = val;                   \
          else if (OUT_MODE == 1) ((float*)Outp)[o] += val;                   \
          else                    ((__bf16*)Outp)[o] = f32_to_bf16(val);      \
        } } }
  EPI(0, acc0) EPI(1, acc1) EPI(2, acc2) EPI(3, acc3)
  #undef EPI
}

// ---------------------------------------------------------------------------
// RMSNorm: one wave per 128-wide row, 8 rows per 256-thread block.
// ---------------------------------------------------------------------------
__global__ void __launch_bounds__(256)
rms_kernel(const float* __restrict__ h, const float* __restrict__ w,
           __bf16* __restrict__ xn)
{
  const int wv = threadIdx.x >> 5, lane = threadIdx.x & 31;
  const size_t row = (size_t)blockIdx.x * 8 + wv;
  const float* hr = h + row * DMODEL;
  float vals[4];
  float ss = 0.f;
  #pragma unroll
  for (int i = 0; i < 4; ++i) { vals[i] = hr[lane + 32 * i]; ss += vals[i] * vals[i]; }
  #pragma unroll
  for (int off = 16; off; off >>= 1) ss += __shfl_xor(ss, off, 32);
  const float sc = rsqrtf(ss * (1.f / DMODEL) + 1e-5f);
  #pragma unroll
  for (int i = 0; i < 4; ++i)
    xn[row * DMODEL + lane + 32 * i] = f32_to_bf16(vals[i] * sc * w[lane + 32 * i]);
}

// ---------------------------------------------------------------------------
// Depthwise causal conv (DCONV=4) + SiLU, reads xc-half of xz (bf16).
// ---------------------------------------------------------------------------
__global__ void __launch_bounds__(256)
conv_kernel(const __bf16* __restrict__ xz, const float* __restrict__ cw,
            const float* __restrict__ cb, __bf16* __restrict__ xc)
{
  const size_t idx = (size_t)blockIdx.x * 256 + threadIdx.x;
  const int c = (int)(idx & 255);
  const size_t rt = idx >> 8;                 // seq*240 + t
  const int t = (int)(rt % L_SEQ);
  const size_t seqBase = (rt - (size_t)t) * (2 * DINNER);
  float acc = cb[c];
  #pragma unroll
  for (int k = 0; k < DCONV; ++k) {
    const int tt = t - (DCONV - 1) + k;
    if (tt >= 0)
      acc += bf16_to_f32(xz[seqBase + (size_t)tt * (2 * DINNER) + c]) * cw[k * DINNER + c];
  }
  const float s = acc / (1.f + __expf(-acc));  // silu
  xc[rt * DINNER + c] = f32_to_bf16(s);
}

// ---------------------------------------------------------------------------
// Selective scan: block = one sequence, thread = one channel (256).
// xdb (240x40) staged in LDS; 16-state recurrence kept in registers.
// Writes y = (scan + Dp*xc) * silu(z) as bf16 into the xc-half of xz.
// ---------------------------------------------------------------------------
__global__ void __launch_bounds__(256)
scan_kernel(const float* __restrict__ xdb, const __bf16* __restrict__ xc,
            const __bf16* xz, __bf16* ybuf,
            const float* __restrict__ W_dt, const float* __restrict__ b_dt,
            const float* __restrict__ A_log, const float* __restrict__ Dp)
{
  __shared__ float xl[L_SEQ * DXDB];          // 38400 B
  const int c = threadIdx.x;
  const int seq = blockIdx.x;
  for (int e = c; e < L_SEQ * DXDB; e += 256)
    xl[e] = xdb[(size_t)seq * (L_SEQ * DXDB) + e];
  __syncthreads();

  float Aexp[DSTATE];
  #pragma unroll
  for (int s = 0; s < DSTATE; ++s) Aexp[s] = -__expf(A_log[c * DSTATE + s]);
  float wdt[DTRANK];
  #pragma unroll
  for (int r = 0; r < DTRANK; ++r) wdt[r] = W_dt[r * DINNER + c];
  const float bdt = b_dt[c];
  const float dp  = Dp[c];

  float h[DSTATE];
  #pragma unroll
  for (int s = 0; s < DSTATE; ++s) h[s] = 0.f;

  const size_t base = (size_t)seq * L_SEQ;
  for (int t = 0; t < L_SEQ; ++t) {
    const float* row = &xl[t * DXDB];
    float pre = bdt;
    #pragma unroll
    for (int r = 0; r < DTRANK; ++r) pre += row[r] * wdt[r];
    const float dt = (pre > 20.f) ? pre : log1pf(__expf(pre));   // softplus
    const float x_t = bf16_to_f32(xc[(base + t) * DINNER + c]);
    const float dtx = dt * x_t;
    float y = 0.f;
    #pragma unroll
    for (int s = 0; s < DSTATE; ++s) {
      const float dA = __expf(dt * Aexp[s]);
      h[s] = h[s] * dA + dtx * row[DTRANK + s];          // B_t
      y += h[s] * row[DTRANK + DSTATE + s];              // C_t
    }
    const float z = bf16_to_f32(xz[(base + t) * (2 * DINNER) + DINNER + c]);
    const float g = z / (1.f + __expf(-z));              // silu(z)
    ybuf[(base + t) * (2 * DINNER) + c] = f32_to_bf16((y + dp * x_t) * g);
  }
}

// ---------------------------------------------------------------------------
// Head: final rmsnorm on last timestep + 128->64 GELU -> 1. Block = 128.
// ---------------------------------------------------------------------------
__global__ void __launch_bounds__(128)
head_kernel(const float* __restrict__ hres, const float* __restrict__ fw,
            const float* __restrict__ W1, const float* __restrict__ b1,
            const float* __restrict__ W2, const float* __restrict__ b2,
            float* __restrict__ out)
{
  __shared__ float hn[DMODEL];
  __shared__ float red[4];
  __shared__ float red2[2];
  const int tid = threadIdx.x;
  const int seq = blockIdx.x;
  const float v = hres[((size_t)seq * L_SEQ + (L_SEQ - 1)) * DMODEL + tid];
  float ss = v * v;
  #pragma unroll
  for (int off = 16; off; off >>= 1) ss += __shfl_xor(ss, off, 32);
  if ((tid & 31) == 0) red[tid >> 5] = ss;
  __syncthreads();
  const float tot = red[0] + red[1] + red[2] + red[3];
  const float sc = rsqrtf(tot * (1.f / DMODEL) + 1e-5f);
  hn[tid] = v * sc * fw[tid];
  __syncthreads();
  float part = 0.f;
  if (tid < 64) {
    float acc = b1[tid];
    #pragma unroll 4
    for (int k = 0; k < DMODEL; ++k) acc += hn[k] * W1[k * 64 + tid];
    const float g = 0.5f * acc * (1.f + erff(acc * 0.70710678118f));   // exact GELU
    part = g * W2[tid];
  }
  #pragma unroll
  for (int off = 16; off; off >>= 1) part += __shfl_xor(part, off, 32);
  if (tid < 64 && (tid & 31) == 0) red2[tid >> 5] = part;
  __syncthreads();
  if (tid == 0) out[seq] = red2[0] + red2[1] + b2[0];
}

// ---------------------------------------------------------------------------
extern "C" void kernel_launch(void* const* d_in, const int* in_sizes, int n_in,
                              void* d_out, int out_size, void* d_ws, size_t ws_size,
                              hipStream_t stream)
{
  (void)in_sizes; (void)n_in; (void)out_size; (void)ws_size;
  const float* x       = (const float*)d_in[0];
  const float* W_enc   = (const float*)d_in[1];
  const float* b_enc   = (const float*)d_in[2];
  const float* norm_w  = (const float*)d_in[3];
  const float* W_in    = (const float*)d_in[4];
  const float* conv_w  = (const float*)d_in[5];
  const float* conv_b  = (const float*)d_in[6];
  const float* W_xp    = (const float*)d_in[7];
  const float* W_dt    = (const float*)d_in[8];
  const float* b_dt    = (const float*)d_in[9];
  const float* A_log   = (const float*)d_in[10];
  const float* Dp      = (const float*)d_in[11];
  const float* W_out   = (const float*)d_in[12];
  const float* final_w = (const float*)d_in[13];
  const float* W_h1    = (const float*)d_in[14];
  const float* b_h1    = (const float*)d_in[15];
  const float* W_h2    = (const float*)d_in[16];
  const float* b_h2    = (const float*)d_in[17];
  float* out = (float*)d_out;

  char* ws = (char*)d_ws;
  float*  hres = (float*)(ws);                       // 58,982,400 B
  __bf16* xn   = (__bf16*)(ws + 58982400);           // 29,491,200 B
  __bf16* xz   = (__bf16*)(ws + 88473600);           // 117,964,800 B
  __bf16* xc   = (__bf16*)(ws + 206438400);          // 58,982,400 B
  float*  xdb  = (float*)(ws + 265420800);           // 18,432,000 B
  __bf16* xe   = (__bf16*)(ws + 283852800);          // 66,355,200 B

  const dim3 blk(256);

  // x -> bf16, zero-padded rows of 288
  cast_pad_kernel<<<dim3(M_ROWS), blk, 0, stream>>>(x, xe);

  // Encoder: hres = xe @ W_enc + b_enc   (Kpad=288 -> 9 k-tiles, KB=258)
  gemm_wmma_kernel<0, true><<<dim3(M_ROWS / 128, 2), blk, 0, stream>>>(
      xe, W_enc, hres, b_enc, 9, F_IN, F_PAD, DMODEL, DMODEL, DMODEL, DMODEL);

  for (int l = 0; l < NLAYERS; ++l) {
    rms_kernel<<<dim3(M_ROWS / 8), blk, 0, stream>>>(hres, norm_w + l * DMODEL, xn);

    // xz = xn @ W_in[l]   (115200 x 128 x 512)
    gemm_wmma_kernel<2, false><<<dim3(M_ROWS / 128, 8), blk, 0, stream>>>(
        xn, W_in + (size_t)l * DMODEL * 2 * DINNER, xz, nullptr,
        4, DMODEL, DMODEL, 2 * DINNER, 2 * DINNER, 2 * DINNER, 2 * DINNER);

    conv_kernel<<<dim3(M_ROWS), blk, 0, stream>>>(
        xz, conv_w + (size_t)l * DCONV * DINNER, conv_b + (size_t)l * DINNER, xc);

    // xdb = xc @ W_xp[l]   (N=40 padded to one 64-wide tile)
    gemm_wmma_kernel<0, false><<<dim3(M_ROWS / 128, 1), blk, 0, stream>>>(
        xc, W_xp + (size_t)l * DINNER * DXDB, xdb, nullptr,
        8, DINNER, DINNER, DXDB, DXDB, DXDB, DXDB);

    // sequential SSM scan; writes gated y into the xc-half of xz (ldA=512)
    scan_kernel<<<dim3(S_SEQ), blk, 0, stream>>>(
        xdb, xc, xz, xz,
        W_dt + (size_t)l * DTRANK * DINNER, b_dt + (size_t)l * DINNER,
        A_log + (size_t)l * DINNER * DSTATE, Dp + (size_t)l * DINNER);

    // hres += y @ W_out[l]
    gemm_wmma_kernel<1, false><<<dim3(M_ROWS / 128, 2), blk, 0, stream>>>(
        xz, W_out + (size_t)l * DINNER * DMODEL, hres, nullptr,
        8, DINNER, 2 * DINNER, DMODEL, DMODEL, DMODEL, DMODEL);
  }

  head_kernel<<<dim3(S_SEQ), dim3(128), 0, stream>>>(
      hres, final_w, W_h1, b_h1, W_h2, b_h2, out);
}
